// SpatialImageLanguageAttention_65326452572510
// MI455X (gfx1250) — matrix operand order repaired and checked by reading, not compiled, not run on hardware
//
#include <hip/hip_runtime.h>

typedef __attribute__((ext_vector_type(16))) __bf16 v16bf;
typedef __attribute__((ext_vector_type(8)))  float  v8f;

#define B_SZ 32
#define HW   1024
#define C    768
#define NL   32
#define NH   8
#define DH   96
#define TN   64
#define XLD  776   // 768 + 8 bf16 pad -> conflict-free A-frag reads
#define KVLD 104   // 96 + 8 bf16 pad
#define NKT  (C / 32)   // 24 K-tiles over the 768 contraction dim
#define NMT  (C / 16)   // 48 M/N-tiles over the 768 output dim

// ---------------- WMMA fragment loaders (wave32, 16x16x32 bf16) ----------------
// A (16Mx32K): lanes0-15 M=lane, elems 0-15->K+0..15;
//              lanes16-31 M=lane-16, elems 0-15->K+16..31.
static __device__ inline v16bf load_a_bf16(const __bf16* base, int ld, int m0, int k0, int lane) {
  v16bf a;
  const int m = lane & 15;
  const int kb = (lane < 16) ? 0 : 16;
  const __bf16* row = base + (m0 + m) * ld + k0;
#pragma unroll
  for (int e = 0; e < 8; ++e) { a[e] = row[kb + e]; a[8 + e] = row[kb + 8 + e]; }
  return a;
}
// A[M][K] where memory holds base[K*ld + M] (transposed f32 source)
static __device__ inline v16bf load_aT_f32(const float* base, int ld, int m0, int k0, int lane) {
  v16bf a;
  const int m = lane & 15;
  const int kb = (lane < 16) ? 0 : 16;
#pragma unroll
  for (int e = 0; e < 8; ++e) {
    a[e]     = (__bf16)base[(k0 + kb + e) * ld + m0 + m];
    a[8 + e] = (__bf16)base[(k0 + kb + 8 + e) * ld + m0 + m];
  }
  return a;
}
// B (32Kx16N): N=lane&15; lanes0-15 K=k0..k0+15, lanes16-31 K=k0+16..31.
static __device__ inline v16bf load_b_bf16(const __bf16* base, int ld, int k0, int n0, int lane) {
  v16bf b;
  const int n = lane & 15;
  const int kb = (lane < 16) ? 0 : 16;
#pragma unroll
  for (int e = 0; e < 16; ++e) b[e] = base[(k0 + kb + e) * ld + n0 + n];
  return b;
}
// B[K][N] where memory holds base[N*ld + K] (transposed f32 source, contiguous in K)
static __device__ inline v16bf load_bT_f32(const float* base, int ld, int k0, int n0, int lane) {
  v16bf b;
  const int n = lane & 15;
  const int kb = (lane < 16) ? 0 : 16;
  const float* row = base + (size_t)(n0 + n) * ld + k0 + kb;
#pragma unroll
  for (int e = 0; e < 16; ++e) b[e] = (__bf16)row[e];
  return b;
}
// Fragment fetch from pre-packed layout: 32 contiguous bytes per lane.
static __device__ inline v16bf fetch_frag(const __bf16* pack, int mt, int kt, int lane) {
  return *(const v16bf*)(pack + (((size_t)mt * NKT + kt) * 32 + lane) * 16);
}
static __device__ inline v8f wmma32(v16bf a, v16bf b, v8f c) {
  return __builtin_amdgcn_wmma_f32_16x16x32_bf16(false, a, false, b, (short)0, c, false, false);
}
static __device__ inline unsigned short bf_bits(float f) {
  __bf16 h = (__bf16)f;
  return __builtin_bit_cast(unsigned short, h);
}

// ---------------- Kernel 0: pack 768x768 f32 weight into bf16 WMMA fragments ----------------
// amode=1: A-matrix pattern (W rows are M, cols are K).
// amode=0: B-matrix pattern (W rows are N, cols are K; i.e. B[K][N] = W[N][K]).
__global__ __launch_bounds__(256) void pack_w_kernel(const float* __restrict__ W,
                                                     __bf16* __restrict__ out, int amode) {
  const int t = blockIdx.x * 256 + threadIdx.x;
  const int tile = t >> 5, lane = t & 31;
  if (tile >= NMT * NKT) return;
  const int mt = tile / NKT, kt = tile % NKT;
  const int row = mt * 16 + (lane & 15);
  const float* wrow = W + (size_t)row * C + kt * 32;
  __bf16* dst = out + ((size_t)tile * 32 + lane) * 16;
  if (amode) {
    const int kb = (lane < 16) ? 0 : 16;
#pragma unroll
    for (int e = 0; e < 8; ++e) {
      dst[e]     = (__bf16)wrow[kb + e];
      dst[8 + e] = (__bf16)wrow[kb + 8 + e];
    }
  } else {
    const int kb = (lane < 16) ? 0 : 16;
#pragma unroll
    for (int e = 0; e < 16; ++e) dst[e] = (__bf16)wrow[kb + e];
  }
}

// ---------------- Kernel 1: q = IN(wq @ l + bq) * mask -> q_pack (B-frag layout per head) ----
// q_pack layout: [b][h][kt(3)][lt(2)][lane(32)][e(16)] bf16
__global__ __launch_bounds__(256) void qproj_kernel(
    const float* __restrict__ l, const float* __restrict__ mask,
    const __bf16* __restrict__ wq_pack, const float* __restrict__ bq,
    __bf16* __restrict__ q_pack) {
  __shared__ __align__(32) __bf16 l_pack[NKT * 2 * 32 * 16];  // B-frag layout of l[b]
  __shared__ float scratch[8][16][NL];
  __shared__ float m_lds[NL];
  const int b = blockIdx.x, tid = threadIdx.x;
  const int wave = tid >> 5, lane = tid & 31;
  const float* lb = l + (size_t)b * C * NL;
  // stage l[b] (c-major rows of 32) into B-fragment layout
  for (int i = tid; i < C * NL; i += 256) {
    const int c = i >> 5, li = i & 31;
    const int kt = c >> 5, krel = c & 31;
    const int laneoff = (krel >= 16) ? 16 : 0, e = krel & 15;
    const int lt = li >> 4, n = li & 15;
    l_pack[(((kt * 2 + lt) * 32) + n + laneoff) * 16 + e] = (__bf16)lb[i];
  }
  if (tid < NL) m_lds[tid] = mask[b * NL + tid];
  __syncthreads();

  for (int mt = wave; mt < NMT; mt += 8) {
    const int k0 = mt * 16;
    v8f acc0 = {}, acc1 = {};
    for (int kt = 0; kt < NKT; ++kt) {
      v16bf a  = fetch_frag(wq_pack, mt, kt, lane);
      v16bf b0 = *(const v16bf*)(l_pack + ((kt * 2 + 0) * 32 + lane) * 16);
      v16bf b1 = *(const v16bf*)(l_pack + ((kt * 2 + 1) * 32 + lane) * 16);
      acc0 = wmma32(a, b0, acc0);
      acc1 = wmma32(a, b1, acc1);
    }
    const int r0 = (lane < 16) ? 0 : 8, n = lane & 15;
#pragma unroll
    for (int r = 0; r < 8; ++r) {
      scratch[wave][r + r0][n]      = acc0[r];
      scratch[wave][r + r0][16 + n] = acc1[r];
    }
    __syncthreads();
    if (lane < 16) {
      const int k = k0 + lane;
      const float bias = bq[k];
      float vals[NL], s = 0.f, sq = 0.f;
#pragma unroll
      for (int li = 0; li < NL; ++li) {
        float v = scratch[wave][lane][li] + bias;
        vals[li] = v; s += v; sq += v * v;
      }
      const float mu = s * (1.f / NL);
      const float var = sq * (1.f / NL) - mu * mu;
      const float rs = rsqrtf(var + 1e-5f);
      // scatter into q_pack fragment layout
      const int h = k / DH, drel = k % DH;
      const int kt = drel >> 5, krel = drel & 31;
      const int laneoff = (krel >= 16) ? 16 : 0, e = krel & 15;
      __bf16* qh = q_pack + ((((size_t)b * NH + h) * 3 + kt) * 2) * 32 * 16;
#pragma unroll
      for (int li = 0; li < NL; ++li) {
        const int lt = li >> 4, n2 = li & 15;
        qh[((size_t)lt * 32 + n2 + laneoff) * 16 + e] =
            (__bf16)((vals[li] - mu) * rs * m_lds[li]);
      }
    }
    __syncthreads();
  }
}

// ---------------- Kernel 2: fused K/V proj + attention, per (b, 64-row n-tile) ----------------
__global__ __launch_bounds__(256) void attn_kernel(
    const float* __restrict__ x, const float* __restrict__ mask,
    const __bf16* __restrict__ wk_pack, const float* __restrict__ bk,
    const __bf16* __restrict__ wv_pack, const float* __restrict__ bv,
    const __bf16* __restrict__ q_pack, float* __restrict__ out_acc) {
  __shared__ __align__(32) __bf16 x_lds[TN * XLD];
  __shared__ __align__(32) __bf16 kv_lds[TN * KVLD];
  __shared__ float s_lds[TN * NL];
  __shared__ float m_lds[NL];
  const int b = blockIdx.y, n0 = blockIdx.x * TN;
  const int tid = threadIdx.x, wave = tid >> 5, lane = tid & 31;
  const float simscale = 0.03608439182435161f;  // 768^-0.5

  // stage x tile as bf16 via float4 loads + packed 8-byte LDS stores
  const float4* xb4 = (const float4*)(x + ((size_t)b * HW + n0) * C);
  for (int i = tid; i < TN * (C / 4); i += 256) {
    const int r = i / (C / 4), c4 = i % (C / 4);
    float4 v = xb4[i];
    uint2 p;
    p.x = (unsigned)bf_bits(v.x) | ((unsigned)bf_bits(v.y) << 16);
    p.y = (unsigned)bf_bits(v.z) | ((unsigned)bf_bits(v.w) << 16);
    *(uint2*)&x_lds[r * XLD + c4 * 4] = p;
  }
  if (tid < NL) m_lds[tid] = mask[b * NL + tid];
  __syncthreads();

  for (int h = 0; h < NH; ++h) {
    // ---- K projection: 64n x 96d tile; wave w: Mtile = w&3, Ntiles = (w>>2)+2j ----
    {
      const int mi = wave & 3, nj0 = wave >> 2;
      v8f kacc[3] = {{}, {}, {}};
      for (int kt = 0; kt < NKT; ++kt) {
        v16bf a = load_a_bf16(x_lds, XLD, mi * 16, kt * 32, lane);
#pragma unroll
        for (int j = 0; j < 3; ++j) {
          const int ni = nj0 + 2 * j;
          v16bf bf = fetch_frag(wk_pack, h * 6 + ni, kt, lane);
          kacc[j] = wmma32(a, bf, kacc[j]);
        }
      }
      const int r0 = (lane < 16) ? 0 : 8, n = lane & 15;
#pragma unroll
      for (int j = 0; j < 3; ++j) {
        const int ni = nj0 + 2 * j;
        const float bias = bk[h * DH + ni * 16 + n];
#pragma unroll
        for (int r = 0; r < 8; ++r)
          kv_lds[(mi * 16 + r + r0) * KVLD + ni * 16 + n] = (__bf16)(kacc[j][r] + bias);
      }
    }
    __syncthreads();

    // ---- sim: 64n x 32l; one 16x16 tile per wave ----
    {
      const int mi = wave & 3, ni = wave >> 2;
      v8f sacc = {};
      const __bf16* qh = q_pack + ((((size_t)b * NH + h) * 3) * 2) * 32 * 16;
#pragma unroll
      for (int kt = 0; kt < 3; ++kt) {
        v16bf a  = load_a_bf16(kv_lds, KVLD, mi * 16, kt * 32, lane);
        v16bf bf = *(const v16bf*)(qh + ((kt * 2 + ni) * 32 + lane) * 16);
        sacc = wmma32(a, bf, sacc);
      }
      const int r0 = (lane < 16) ? 0 : 8, n = lane & 15;
#pragma unroll
      for (int r = 0; r < 8; ++r)
        s_lds[(mi * 16 + r + r0) * NL + ni * 16 + n] = sacc[r] * simscale;
    }
    __syncthreads();

    // ---- softmax over l (rows local) ----
    if (tid < TN) {
      float* row = s_lds + tid * NL;
      float v2[NL], mx = -1e30f;
#pragma unroll
      for (int li = 0; li < NL; ++li) {
        float xv = row[li] + 10000.f * (m_lds[li] - 1.f);
        v2[li] = xv; mx = fmaxf(mx, xv);
      }
      float sum = 0.f;
#pragma unroll
      for (int li = 0; li < NL; ++li) { float e = __expf(v2[li] - mx); v2[li] = e; sum += e; }
      const float inv = 1.f / sum;
#pragma unroll
      for (int li = 0; li < NL; ++li) row[li] = v2[li] * inv;
    }

    // ---- V projection into kv_lds (free after sim barrier) ----
    {
      const int mi = wave & 3, nj0 = wave >> 2;
      v8f vacc[3] = {{}, {}, {}};
      for (int kt = 0; kt < NKT; ++kt) {
        v16bf a = load_a_bf16(x_lds, XLD, mi * 16, kt * 32, lane);
#pragma unroll
        for (int j = 0; j < 3; ++j) {
          const int ni = nj0 + 2 * j;
          v16bf bf = fetch_frag(wv_pack, h * 6 + ni, kt, lane);
          vacc[j] = wmma32(a, bf, vacc[j]);
        }
      }
      const int r0 = (lane < 16) ? 0 : 8, n = lane & 15;
#pragma unroll
      for (int j = 0; j < 3; ++j) {
        const int ni = nj0 + 2 * j;
        const float bias = bv[h * DH + ni * 16 + n];
#pragma unroll
        for (int r = 0; r < 8; ++r)
          kv_lds[(mi * 16 + r + r0) * KVLD + ni * 16 + n] = (__bf16)(vacc[j][r] + bias);
      }
    }
    __syncthreads();

    // ---- out_h[l=32][d=96] += P^T(32x64) @ V(64x96); 12 tiles over 8 waves ----
    for (int t = wave; t < 12; t += 8) {
      const int mi = t & 1, ni = t >> 1;
      v8f oacc = {};
#pragma unroll
      for (int nb = 0; nb < TN; nb += 32) {
        v16bf a  = load_aT_f32(s_lds, NL, mi * 16, nb, lane);   // A[l][n] = P[n][l]
        v16bf bf = load_b_bf16(kv_lds, KVLD, nb, ni * 16, lane);
        oacc = wmma32(a, bf, oacc);
      }
      const int r0 = (lane < 16) ? 0 : 8, n = lane & 15;
#pragma unroll
      for (int r = 0; r < 8; ++r) {
        const int lrow = mi * 16 + r + r0;
        const int dcol = h * DH + ni * 16 + n;
        atomicAdd(&out_acc[((size_t)b * NL + lrow) * C + dcol], oacc[r]);
      }
    }
    __syncthreads();  // before next head overwrites kv_lds / s_lds
  }
}

// ---------------- Kernel 3: out = IN(wo @ acc^T + bo), transposed store ----------------
__global__ __launch_bounds__(256) void oproj_kernel(
    const __bf16* __restrict__ wo_pack, const float* __restrict__ bo,
    const float* __restrict__ acc, float* __restrict__ out) {
  __shared__ float s_lds[64 * NL];
  const int b = blockIdx.y, o0 = blockIdx.x * 64;
  const int tid = threadIdx.x, wave = tid >> 5, lane = tid & 31;
  const float* accb = acc + (size_t)b * NL * C;

  const int mi = wave & 3, ni = wave >> 2;
  v8f cacc = {};
  for (int kt = 0; kt < NKT; ++kt) {
    v16bf a  = fetch_frag(wo_pack, o0 / 16 + mi, kt, lane);
    v16bf bf = load_bT_f32(accb, C, kt * 32, ni * 16, lane);
    cacc = wmma32(a, bf, cacc);
  }
  const int r0 = (lane < 16) ? 0 : 8, n = lane & 15;
#pragma unroll
  for (int r = 0; r < 8; ++r)
    s_lds[(mi * 16 + r + r0) * NL + ni * 16 + n] = cacc[r];
  __syncthreads();

  if (tid < 64) {
    const int o = o0 + tid;
    const float bias = bo[o];
    float vals[NL], s = 0.f, sq = 0.f;
#pragma unroll
    for (int li = 0; li < NL; ++li) {
      float v = s_lds[tid * NL + li] + bias;
      vals[li] = v; s += v; sq += v * v;
    }
    const float mu = s * (1.f / NL);
    const float var = sq * (1.f / NL) - mu * mu;
    const float rs = rsqrtf(var + 1e-5f);
#pragma unroll
    for (int li = 0; li < NL; ++li)
      out[((size_t)b * NL + li) * C + o] = (vals[li] - mu) * rs;
  }
}

// ---------------- Launch ----------------
extern "C" void kernel_launch(void* const* d_in, const int* in_sizes, int n_in,
                              void* d_out, int out_size, void* d_ws, size_t ws_size,
                              hipStream_t stream) {
  (void)in_sizes; (void)n_in; (void)out_size; (void)ws_size;
  const float* x    = (const float*)d_in[0];
  const float* l    = (const float*)d_in[1];
  const float* mask = (const float*)d_in[2];
  const float* wq   = (const float*)d_in[3];
  const float* bq   = (const float*)d_in[4];
  const float* wk   = (const float*)d_in[5];
  const float* bk   = (const float*)d_in[6];
  const float* wv   = (const float*)d_in[7];
  const float* bv   = (const float*)d_in[8];
  const float* wo   = (const float*)d_in[9];
  const float* bo   = (const float*)d_in[10];

  const size_t WSZ = (size_t)C * C * sizeof(__bf16);  // one packed weight
  char* ws = (char*)d_ws;
  __bf16* wq_pack = (__bf16*)(ws);
  __bf16* wk_pack = (__bf16*)(ws + WSZ);
  __bf16* wv_pack = (__bf16*)(ws + 2 * WSZ);
  __bf16* wo_pack = (__bf16*)(ws + 3 * WSZ);
  __bf16* q_pack  = (__bf16*)(ws + 4 * WSZ);
  float*  out_acc = (float*)(ws + 4 * WSZ + (size_t)B_SZ * C * NL * sizeof(__bf16));

  const int pack_blocks = (NMT * NKT * 32 + 255) / 256;
  pack_w_kernel<<<pack_blocks, 256, 0, stream>>>(wq, wq_pack, 1);
  pack_w_kernel<<<pack_blocks, 256, 0, stream>>>(wk, wk_pack, 0);
  pack_w_kernel<<<pack_blocks, 256, 0, stream>>>(wv, wv_pack, 0);
  pack_w_kernel<<<pack_blocks, 256, 0, stream>>>(wo, wo_pack, 1);
  hipMemsetAsync(out_acc, 0, (size_t)B_SZ * NL * C * sizeof(float), stream);

  qproj_kernel<<<B_SZ, 256, 0, stream>>>(l, mask, wq_pack, bq, q_pack);
  attn_kernel<<<dim3(HW / TN, B_SZ), 256, 0, stream>>>(x, mask, wk_pack, bk, wv_pack, bv,
                                                       q_pack, out_acc);
  oproj_kernel<<<dim3(C / 64, B_SZ), 256, 0, stream>>>(wo_pack, bo, out_acc, (float*)d_out);
}
